// SparseAttention_2482491097600
// MI455X (gfx1250) — compile-verified, matrix operand order checked by reading, not run on hardware
//
#include <hip/hip_runtime.h>

// Problem constants (from reference): B=2, T=8192, C=1024, H=64, window d in [0,16]
#define B_DIM 2
#define T_DIM 8192
#define C_DIM 1024
#define H_DIM 64

#define ROWS 64          // queries per workgroup
#define EXT  80          // ROWS + 16 lookback rows for K/V
#define WAVES 8
#define THREADS 256
#define KC   64          // C-dimension chunk staged per iteration
#define NJOBS 15         // (EXT/16) mtiles * 3 matrices
#define MAXJ 2           // jobs per wave (statically unrolled; job 15 is a dummy)

#define NXLD 5           // EXT*(KC/4)/THREADS  = 1280/256
#define NWLD 12          // 3*KC*(H_DIM/4)/THREADS = 3072/256

// LDS strides in halves (all row bases 16B-aligned)
#define XS  72           // x chunk row:    KC + 8 pad
#define WTS 72           // wT row (c-dim): KC + 8 pad
#define QS  72           // q/k staging row: H + 8 pad
#define VTS 88           // vT row (ext rows): EXT + 8 pad
#define PS  40           // prob row: 32 keys + 8 pad

// Region A (phase-1 streaming) and Region B (staging/phase-2) overlap in LDS.
#define OFF_XS 0
#define OFF_WT (EXT*XS)                       // 5760
#define REGION_A (EXT*XS + 3*H_DIM*WTS)       // 19584 halves

#define OFF_QS 0
#define OFF_KS (EXT*QS)                       // 5760
#define OFF_VT (2*EXT*QS)                     // 11520
#define OFF_PS (2*EXT*QS + H_DIM*VTS)         // 17152
#define REGION_B (OFF_PS + WAVES*16*PS)       // 22272 halves

#define SMEM_HALVES (REGION_A > REGION_B ? REGION_A : REGION_B)

typedef __attribute__((ext_vector_type(16))) _Float16 v16h;
typedef __attribute__((ext_vector_type(8)))  _Float16 v8h;
typedef __attribute__((ext_vector_type(4)))  _Float16 v4h;
typedef __attribute__((ext_vector_type(8)))  float    v8f;

// Build a WMMA 16-half fragment from LDS: halves [p..p+7] and [p+16..p+23].
// Matches the CDNA5 16-bit A/B lane layout when p = rowbase + kk + (lane>=16 ? 8 : 0).
__device__ __forceinline__ v16h ld_frag(const _Float16* p) {
  v8h lo = *(const v8h*)(p);
  v8h hi = *(const v8h*)(p + 16);
  return __builtin_shufflevector(lo, hi, 0,1,2,3,4,5,6,7,8,9,10,11,12,13,14,15);
}

__global__ __launch_bounds__(THREADS) void sparse_attn_fused(
    const float* __restrict__ x,  const float* __restrict__ Wq,
    const float* __restrict__ Wk, const float* __restrict__ Wv,
    float* __restrict__ out) {
  __shared__ __align__(16) _Float16 smem[SMEM_HALVES];

  const int bb   = blockIdx.y;
  const int t0   = blockIdx.x * ROWS;          // first query row of this WG
  const int lane = threadIdx.x & 31;
  const int wave = threadIdx.x >> 5;
  const int hs   = (lane >> 4);                // 0: lanes 0-15, 1: lanes 16-31
  const int l15  = lane & 15;
  const int hoff = hs ? 8 : 0;

  _Float16* xs = smem + OFF_XS;                // [EXT][XS] f16 x chunk
  _Float16* wT = smem + OFF_WT;                // [3][H][WTS] f16 W chunk, transposed

  // Static job assignment: wave w owns jobs {w, w+8}; job 15 is a dummy (clamped,
  // computed to keep control flow uniform, never staged).
  int mats[MAXJ], mtiles[MAXJ], a_base[MAXJ], w_base[MAXJ];
  bool jvalid[MAXJ];
#pragma unroll
  for (int s = 0; s < MAXJ; ++s) {
    const int job = wave + s*WAVES;
    jvalid[s] = (job < NJOBS);
    const int jb = jvalid[s] ? job : 0;
    mats[s]   = jb % 3;
    mtiles[s] = jb / 3;
    a_base[s] = (mtiles[s]*16 + l15)*XS;
    w_base[s] = (mats[s]*H_DIM + l15)*WTS;
  }

  v8f zf = {};
  v8f acc[MAXJ][4];
#pragma unroll
  for (int s = 0; s < MAXJ; ++s)
#pragma unroll
    for (int n = 0; n < 4; ++n) acc[s][n] = zf;

  // ---------------- Phase 1: Q/K/V projections, streamed over C ----------------
  // Double-buffered through registers: global loads for chunk c+1 are issued
  // before the WMMA section of chunk c, hiding HBM latency behind compute.
  float4 xr[NXLD];
  float4 wr[NWLD];

  auto prefetch = [&](int c0) {
#pragma unroll
    for (int k = 0; k < NXLD; ++k) {
      const int i  = threadIdx.x + k*THREADS;
      const int r  = i >> 4;                 // KC/4 == 16
      const int cq = (i & 15) << 2;
      int t = t0 - 16 + r; if (t < 0) t = 0;
      xr[k] = *(const float4*)(x + ((size_t)bb*T_DIM + t)*C_DIM + c0 + cq);
    }
#pragma unroll
    for (int k = 0; k < NWLD; ++k) {
      const int i   = threadIdx.x + k*THREADS;
      const int m   = i >> 10;               // KC*(H_DIM/4) == 1024
      const int rem = i & 1023;
      const int c   = rem >> 4;
      const int h4  = (rem & 15) << 2;
      const float* wsrc = (m == 0) ? Wq : ((m == 1) ? Wk : Wv);
      wr[k] = *(const float4*)(wsrc + (size_t)(c0 + c)*H_DIM + h4);
    }
  };

  auto commit = [&]() {
#pragma unroll
    for (int k = 0; k < NXLD; ++k) {
      const int i  = threadIdx.x + k*THREADS;
      const int r  = i >> 4;
      const int cq = (i & 15) << 2;
      v4h hv;
      hv[0] = (_Float16)xr[k].x; hv[1] = (_Float16)xr[k].y;
      hv[2] = (_Float16)xr[k].z; hv[3] = (_Float16)xr[k].w;
      *(v4h*)(xs + r*XS + cq) = hv;          // one ds_store_b64
    }
#pragma unroll
    for (int k = 0; k < NWLD; ++k) {
      const int i   = threadIdx.x + k*THREADS;
      const int m   = i >> 10;
      const int rem = i & 1023;
      const int c   = rem >> 4;
      const int h4  = (rem & 15) << 2;
      _Float16* b = wT + (m*H_DIM)*WTS + c;  // transposed: strided b16 stores
      b[(h4+0)*WTS] = (_Float16)wr[k].x; b[(h4+1)*WTS] = (_Float16)wr[k].y;
      b[(h4+2)*WTS] = (_Float16)wr[k].z; b[(h4+3)*WTS] = (_Float16)wr[k].w;
    }
  };

  prefetch(0);
  for (int c0 = 0; c0 < C_DIM; c0 += KC) {
    __syncthreads();   // previous chunk's WMMA reads of xs/wT complete
    commit();
    __syncthreads();
    if (c0 + KC < C_DIM) prefetch(c0 + KC);  // in flight during WMMAs below

#pragma unroll
    for (int s = 0; s < MAXJ; ++s) {
      v16h a0 = ld_frag(xs + a_base[s] + 0  + hoff);
      v16h a1 = ld_frag(xs + a_base[s] + 32 + hoff);
#pragma unroll
      for (int nt = 0; nt < 4; ++nt) {
        v16h b0 = ld_frag(wT + w_base[s] + nt*16*WTS + 0  + hoff);
        v16h b1 = ld_frag(wT + w_base[s] + nt*16*WTS + 32 + hoff);
        acc[s][nt] = __builtin_amdgcn_wmma_f32_16x16x32_f16(
            false, a0, false, b0, (short)0, acc[s][nt], false, false);
        acc[s][nt] = __builtin_amdgcn_wmma_f32_16x16x32_f16(
            false, a1, false, b1, (short)0, acc[s][nt], false, false);
      }
    }
  }

  __syncthreads();   // all waves done with xs/wT; region B may overwrite region A

  // Stage Q/K row-major and V transposed, all f16.
  _Float16* qs = smem + OFF_QS;                 // [EXT][QS]
  _Float16* ks = smem + OFF_KS;                 // [EXT][QS]
  _Float16* vT = smem + OFF_VT;                 // [H][VTS]
  _Float16* ps = smem + OFF_PS + wave*16*PS;    // per-wave [16][PS]

#pragma unroll
  for (int s = 0; s < MAXJ; ++s) {
    if (jvalid[s]) {
      const int mat = mats[s];
#pragma unroll
      for (int nt = 0; nt < 4; ++nt) {
#pragma unroll
        for (int r = 0; r < 8; ++r) {
          const int mr = mtiles[s]*16 + r + hs*8;  // C/D layout: M = r (+8 hi half)
          const int hc = nt*16 + l15;              // N = lane%16
          const _Float16 hv = (_Float16)acc[s][nt][r];
          if      (mat == 0) qs[mr*QS + hc] = hv;
          else if (mat == 1) ks[mr*QS + hc] = hv;
          else               vT[hc*VTS + mr] = hv;
        }
      }
    }
  }
  __syncthreads();

  // ---------------- Phase 2: banded attention ----------------
  // 8 waves, 4 query tiles: wave pair (tile, tile+4) redundantly computes the
  // cheap score/softmax part, then each wave produces 2 of the 4 output n-tiles.
  const int tile  = wave & 3;
  const int half  = wave >> 2;        // 0: h columns 0..31, 1: h columns 32..63
  const int qbase = 16 + tile*16;     // ext-row of first query of this tile
  const int kbase = tile*16;          // ext-row of first key (covers j in [i-16, i])

  v8f sc0 = zf, sc1 = zf;
#pragma unroll
  for (int kk = 0; kk < H_DIM; kk += 32) {
    const int aoff = kk + hoff;
    v16h a  = ld_frag(qs + (qbase + l15)*QS + aoff);
    v16h b0 = ld_frag(ks + (kbase + l15)*QS + aoff);
    v16h b1 = ld_frag(ks + (kbase + 16 + l15)*QS + aoff);
    sc0 = __builtin_amdgcn_wmma_f32_16x16x32_f16(false, a, false, b0, (short)0, sc0, false, false);
    sc1 = __builtin_amdgcn_wmma_f32_16x16x32_f16(false, a, false, b1, (short)0, sc1, false, false);
  }

  const float scale = 0.125f;   // 1/sqrt(H)
#pragma unroll
  for (int r = 0; r < 8; ++r) {
    const int m = r + hs*8;     // query local index
    const int n = l15;          // key local index within block
    float s0 = sc0[r] * scale;  // keys kbase+n    -> d = 16 + m - n
    float s1 = sc1[r] * scale;  // keys kbase+16+n -> d = m - n
    const int j0 = t0 - 16 + kbase + n;
    const bool k0 = (m <= n) && (j0 >= 0);     // 1 <= d <= 16, valid key
    const bool k1 = (n <= m);                  // 0 <= d <= 15
    s0 = k0 ? s0 : -3.0e38f;
    s1 = k1 ? s1 : -3.0e38f;
    float mx = fmaxf(s0, s1);
#pragma unroll
    for (int d = 1; d < 16; d <<= 1) mx = fmaxf(mx, __shfl_xor(mx, d, 32));
    float e0 = k0 ? __expf(s0 - mx) : 0.0f;
    float e1 = k1 ? __expf(s1 - mx) : 0.0f;
    float sm = e0 + e1;
#pragma unroll
    for (int d = 1; d < 16; d <<= 1) sm += __shfl_xor(sm, d, 32);
    const float inv = 1.0f / sm;    // diag element always kept -> sm > 0
    ps[m*PS + n]      = (_Float16)(e0 * inv);
    ps[m*PS + 16 + n] = (_Float16)(e1 * inv);
  }
  __syncthreads();

  // out tile = P[16x32] * V[32x32-half] via 2 WMMAs per wave
  v16h pa = ld_frag(ps + l15*PS + hoff);
  v8f o[2];
#pragma unroll
  for (int u = 0; u < 2; ++u) o[u] = zf;
#pragma unroll
  for (int u = 0; u < 2; ++u) {
    const int nt = half*2 + u;
    v16h vb = ld_frag(vT + (nt*16 + l15)*VTS + kbase + hoff);
    o[u] = __builtin_amdgcn_wmma_f32_16x16x32_f16(false, pa, false, vb, (short)0, o[u], false, false);
  }
#pragma unroll
  for (int u = 0; u < 2; ++u) {
    const int nt = half*2 + u;
#pragma unroll
    for (int r = 0; r < 8; ++r) {
      const int m = r + hs*8;
      const int t = t0 + tile*16 + m;
      const int h = nt*16 + l15;
      out[((size_t)bb*T_DIM + t)*H_DIM + h] = o[u][r];
    }
  }
}

extern "C" void kernel_launch(void* const* d_in, const int* in_sizes, int n_in,
                              void* d_out, int out_size, void* d_ws, size_t ws_size,
                              hipStream_t stream) {
  (void)in_sizes; (void)n_in; (void)out_size; (void)d_ws; (void)ws_size;
  const float* x  = (const float*)d_in[0];
  const float* Wq = (const float*)d_in[1];
  const float* Wk = (const float*)d_in[2];
  const float* Wv = (const float*)d_in[3];
  float* out = (float*)d_out;
  dim3 grid(T_DIM / ROWS, B_DIM);
  sparse_attn_fused<<<grid, dim3(THREADS), 0, stream>>>(x, Wq, Wk, Wv, out);
}
